// RNNDemo_85298050498864
// MI455X (gfx1250) — matrix operand order repaired and checked by reading, not compiled, not run on hardware
//
#include <hip/hip_runtime.h>

// CDNA5 / gfx1250, wave32. One wave handles a 16-row batch tile for all S steps.
// Transposed formulation: h'^T = relu(W * h^T + bias + x-term), so weights are the
// WMMA A operand (constant registers) and the recurrent state is the B operand.
// D->B state recycling: cvt to packed bf16, relu via v_pk_max_num_bf16, then one
// v_permlane16_swap per dword pair. Output projection is itself a WMMA with a
// one-row A operand (W_out), so r_out is never materialized.

typedef __attribute__((ext_vector_type(16))) __bf16 v16bf;
typedef __attribute__((ext_vector_type(8)))  float  v8f;
typedef __attribute__((ext_vector_type(2)))  float  f32x2;
typedef __attribute__((ext_vector_type(2)))  __bf16 bf16x2;
typedef __attribute__((ext_vector_type(2)))  unsigned int uint2v;

#define SWAPX16 0x401F  // ds_swizzle group-of-32: xor_mask=0x10, and_mask=0x1f

#if defined(__has_builtin)
#if __has_builtin(__builtin_amdgcn_permlane16_swap)
#define HAVE_PLSWAP 1
#endif
#endif

__device__ __forceinline__ int pack_bf2(float lo, float hi) {
  f32x2 t; t[0] = lo; t[1] = hi;
  bf16x2 b = __builtin_convertvector(t, bf16x2);  // v_cvt_pk_bf16_f32
  union { bf16x2 b; int i; } u; u.b = b;
  return u.i;
}

__device__ __forceinline__ float relu1(float a) {
  // med3(a, 0, +big) == clamp(a, 0, big) == relu, single v_med3 op
  return __builtin_amdgcn_fmed3f(a, 0.0f, 3.38953139e38f);
}

__device__ __forceinline__ int relu_pk_bf16(int a) {
  // packed bf16 max with 0 -> v_pk_max_num_bf16
  union { int i; bf16x2 b; } u; u.i = a;
  bf16x2 z = { (__bf16)0.0f, (__bf16)0.0f };
  u.b = __builtin_elementwise_max(u.b, z);
  return u.i;
}

// A-operand tile (16x32 bf16) of row-major W (32x32): A[m,k] = W[mbase+m, k].
// Contiguous k order (this code's choice): lane m=l&15; element j -> K = j + 16*(l>>4): one 16-float run.
__device__ __forceinline__ v16bf load_wA(const float* W, int mbase, int col, int half) {
  const float* row = W + (mbase + col) * 32 + half * 16;
  float4 a0 = *(const float4*)(row + 0);
  float4 a1 = *(const float4*)(row + 4);
  float4 b0 = *(const float4*)(row + 8);
  float4 b1 = *(const float4*)(row + 12);
  union { v16bf v; int d[8]; } r;
  r.d[0] = pack_bf2(a0.x, a0.y); r.d[1] = pack_bf2(a0.z, a0.w);
  r.d[2] = pack_bf2(a1.x, a1.y); r.d[3] = pack_bf2(a1.z, a1.w);
  r.d[4] = pack_bf2(b0.x, b0.y); r.d[5] = pack_bf2(b0.z, b0.w);
  r.d[6] = pack_bf2(b1.x, b1.y); r.d[7] = pack_bf2(b1.z, b1.w);
  return r.v;
}

// A operand with row 0 = W_out (1x32), rows 1..15 = 0. D row 0 of the projection
// WMMA then holds out[b] for b = lanes 0..15, element 0.
__device__ __forceinline__ v16bf load_woutA(const float* W_out, int col, int half) {
  const float* row = W_out + half * 16;
  float4 a0 = *(const float4*)(row + 0);
  float4 a1 = *(const float4*)(row + 4);
  float4 b0 = *(const float4*)(row + 8);
  float4 b1 = *(const float4*)(row + 12);
  int pk[8];
  pk[0] = pack_bf2(a0.x, a0.y); pk[1] = pack_bf2(a0.z, a0.w);
  pk[2] = pack_bf2(a1.x, a1.y); pk[3] = pack_bf2(a1.z, a1.w);
  pk[4] = pack_bf2(b0.x, b0.y); pk[5] = pack_bf2(b0.z, b0.w);
  pk[6] = pack_bf2(b1.x, b1.y); pk[7] = pack_bf2(b1.z, b1.w);
  union { v16bf v; int d[8]; } r;
#pragma unroll
  for (int u = 0; u < 8; ++u) r.d[u] = (col == 0) ? pk[u] : 0;
  return r.v;
}

// Assemble B operand (32x16: K=hidden, N=batch) from packed-bf16 D-tile dwords.
// B dword u -> K = 16*half + {2u,2u+1}; dword u+4 -> K = 16*half + 8 + {2u,2u+1}.
// v_permlane16_swap_b32(p0, p1) -> ({p0.lo, p1.lo}, {p0.hi, p1.hi}) = (d[u], d[u+4]).
__device__ __forceinline__ v16bf pk_to_B(const int* p0, const int* p1, int half) {
  union { v16bf v; unsigned int d[8]; } r;
#if defined(HAVE_PLSWAP)
  (void)half;
#pragma unroll
  for (int u = 0; u < 4; ++u) {
    uint2v sw = __builtin_amdgcn_permlane16_swap((unsigned)p0[u], (unsigned)p1[u],
                                                 false, false);
    r.d[u]     = sw[0];
    r.d[u + 4] = sw[1];
  }
#else
#pragma unroll
  for (int u = 0; u < 4; ++u) {
    int send = half ? p0[u] : p1[u];
    int recv = __builtin_amdgcn_ds_swizzle(send, SWAPX16);
    r.d[u]     = half ? recv  : p0[u];
    r.d[u + 4] = half ? p1[u] : recv;
  }
#endif
  return r.v;
}

// raw f32 D tiles -> bf16 B operand, no relu (for initial state)
__device__ __forceinline__ v16bf d_to_B(const float* d0, const float* d1, int half) {
  int p0[4], p1[4];
#pragma unroll
  for (int u = 0; u < 4; ++u) {
    p0[u] = pack_bf2(d0[2 * u], d0[2 * u + 1]);
    p1[u] = pack_bf2(d1[2 * u], d1[2 * u + 1]);
  }
  return pk_to_B(p0, p1, half);
}

// raw f32 D tiles -> relu'd bf16 B operand (critical path: cvt, pk_max, swap)
__device__ __forceinline__ v16bf d_to_B_relu(const v8f& d0, const v8f& d1, int half) {
  int p0[4], p1[4];
#pragma unroll
  for (int u = 0; u < 4; ++u) {
    p0[u] = relu_pk_bf16(pack_bf2(d0[2 * u], d0[2 * u + 1]));
    p1[u] = relu_pk_bf16(pack_bf2(d1[2 * u], d1[2 * u + 1]));
  }
  return pk_to_B(p0, p1, half);
}

__device__ __forceinline__ v8f wmma_bf16(v16bf a, v16bf b, v8f c) {
  // D(f32 16x16) = A(bf16 16x32) x B(bf16 32x16) + C
  return __builtin_amdgcn_wmma_f32_16x16x32_bf16(false, a, false, b, (short)0, c,
                                                 false, false);
}

__global__ __launch_bounds__(32)
void rnn2_wmma_kernel(const float* __restrict__ x, const float* __restrict__ h0,
                      const float* __restrict__ W_ih0, const float* __restrict__ W_hh0,
                      const float* __restrict__ b_ih0, const float* __restrict__ b_hh0,
                      const float* __restrict__ W_ih1, const float* __restrict__ W_hh1,
                      const float* __restrict__ b_ih1, const float* __restrict__ b_hh1,
                      const float* __restrict__ W_out, const float* __restrict__ b_out,
                      float* __restrict__ out, int S, int B) {
  const int lane = threadIdx.x;
  const int col  = lane & 15;   // batch index within tile (N for B/D, M row for A)
  const int half = lane >> 4;
  const int bb   = blockIdx.x * 16;

  // ---- constant weight A tiles (fp32 -> bf16 once) ----
  v16bf Whh0_A0 = load_wA(W_hh0,  0, col, half);
  v16bf Whh0_A1 = load_wA(W_hh0, 16, col, half);
  v16bf Wih1_A0 = load_wA(W_ih1,  0, col, half);
  v16bf Wih1_A1 = load_wA(W_ih1, 16, col, half);
  v16bf Whh1_A0 = load_wA(W_hh1,  0, col, half);
  v16bf Whh1_A1 = load_wA(W_hh1, 16, col, half);
  v16bf WoutA   = load_woutA(W_out, col, half);

  // ---- per-lane fp32 fragments of W_ih0 / biases (D-layout: m = 8*half+v) ----
  float wih0_0[8], wih0_1[8], c0_0[8], c0_1[8];
  v8f c1v0, c1v1;
#pragma unroll
  for (int v = 0; v < 8; ++v) {
    int m = half * 8 + v;
    wih0_0[v] = W_ih0[m];
    wih0_1[v] = W_ih0[m + 16];
    c0_0[v] = b_ih0[m]      + b_hh0[m];
    c0_1[v] = b_ih0[m + 16] + b_hh0[m + 16];
    c1v0[v] = b_ih1[m]      + b_hh1[m];
    c1v1[v] = b_ih1[m + 16] + b_hh1[m + 16];
  }
  const float bo = b_out[0];
  const v8f zeroC = {};

  // ---- initial hidden state, raw fp32 D-layout copies + bf16 B operands ----
  float h1f0[8], h1f1[8], h2f0[8], h2f1[8];  // raw pre-relu values
#pragma unroll
  for (int v = 0; v < 8; ++v) {
    int m = half * 8 + v;
    size_t base = (size_t)(bb + col) * 32;
    h1f0[v] = h0[base + m];
    h1f1[v] = h0[base + m + 16];
    h2f0[v] = h0[(size_t)B * 32 + base + m];
    h2f1[v] = h0[(size_t)B * 32 + base + m + 16];
  }
  v16bf h1B = d_to_B(h1f0, h1f1, half);  // initial carry is used un-relu'd
  v16bf h2B = d_to_B(h2f0, h2f1, half);

  const float* xp   = x   + bb;          // x[s*B + bb + col]
  float*       outp = out + bb;          // outs[s*B + bb + col]
  float*       hout = out + (size_t)S * B;

#pragma unroll 2
  for (int s = 0; s < S; ++s) {
    float xv = xp[(size_t)s * B + col];
    // speculative prefetch ahead; translation failures are silently dropped
    __builtin_prefetch(&xp[(size_t)(s + 8) * B + col], 0, 1);

    // ---- layer 0: h1' = relu(W_hh0 * h1^T + x*w_ih0 + b_ih0 + b_hh0) ----
    v8f a0, a1;
#pragma unroll
    for (int v = 0; v < 8; ++v) {
      a0[v] = fmaf(xv, wih0_0[v], c0_0[v]);
      a1[v] = fmaf(xv, wih0_1[v], c0_1[v]);
    }
    a0 = wmma_bf16(Whh0_A0, h1B, a0);
    a1 = wmma_bf16(Whh0_A1, h1B, a1);
    h1B = d_to_B_relu(a0, a1, half);
#pragma unroll
    for (int v = 0; v < 8; ++v) { h1f0[v] = a0[v]; h1f1[v] = a1[v]; }  // raw, off-path

    // ---- layer 1: h2' = relu(W_ih1 * h1'^T + W_hh1 * h2^T + b_ih1 + b_hh1) ----
    v8f d0 = c1v0, d1 = c1v1;
    d0 = wmma_bf16(Wih1_A0, h1B, d0);
    d1 = wmma_bf16(Wih1_A1, h1B, d1);
    d0 = wmma_bf16(Whh1_A0, h2B, d0);
    d1 = wmma_bf16(Whh1_A1, h2B, d1);
    h2B = d_to_B_relu(d0, d1, half);
#pragma unroll
    for (int v = 0; v < 8; ++v) { h2f0[v] = d0[v]; h2f1[v] = d1[v]; }  // raw, off-path

    // ---- output projection as WMMA: row 0 of D = W_out . relu(h2')^T ----
    v8f pr = wmma_bf16(WoutA, h2B, zeroC);
    if (half == 0)
      outp[(size_t)s * B + col] = pr[0] + bo;
  }

  // ---- final hidden state (2,B,32), fp32; deferred relu applied here ----
#pragma unroll
  for (int v = 0; v < 8; ++v) {
    int m = half * 8 + v;
    size_t base = (size_t)(bb + col) * 32;
    hout[base + m]                       = relu1(h1f0[v]);
    hout[base + m + 16]                  = relu1(h1f1[v]);
    hout[(size_t)B * 32 + base + m]      = relu1(h2f0[v]);
    hout[(size_t)B * 32 + base + m + 16] = relu1(h2f1[v]);
  }
}

extern "C" void kernel_launch(void* const* d_in, const int* in_sizes, int n_in,
                              void* d_out, int out_size, void* d_ws, size_t ws_size,
                              hipStream_t stream) {
  const float* x     = (const float*)d_in[0];
  const float* h0    = (const float*)d_in[1];
  const float* W_ih0 = (const float*)d_in[2];
  const float* W_hh0 = (const float*)d_in[3];
  const float* b_ih0 = (const float*)d_in[4];
  const float* b_hh0 = (const float*)d_in[5];
  const float* W_ih1 = (const float*)d_in[6];
  const float* W_hh1 = (const float*)d_in[7];
  const float* b_ih1 = (const float*)d_in[8];
  const float* b_hh1 = (const float*)d_in[9];
  const float* W_out = (const float*)d_in[10];
  const float* b_out = (const float*)d_in[11];
  float* out = (float*)d_out;

  const int B = in_sizes[1] / 64;      // h is (2, B, 32)
  const int S = in_sizes[0] / B;       // x is (S, B, 1)

  dim3 grid(B / 16), block(32);        // 128 waves, one 16-batch tile each
  rnn2_wmma_kernel<<<grid, block, 0, stream>>>(x, h0, W_ih0, W_hh0, b_ih0, b_hh0,
                                               W_ih1, W_hh1, b_ih1, b_hh1,
                                               W_out, b_out, out, S, B);
}
